// my_model3_57131654971882
// MI455X (gfx1250) — compile-verified, matrix operand order checked by reading, not run on hardware
//
#include <hip/hip_runtime.h>
#include <hip/hip_bf16.h>
#include <math.h>

typedef _Float16 f16;
typedef _Float16 v16h __attribute__((ext_vector_type(16)));
typedef _Float16 v8h  __attribute__((ext_vector_type(8)));
typedef float    v8f  __attribute__((ext_vector_type(8)));

#define B_   2
#define L_   1024
#define D_   768
#define H_   12
#define M_   128
#define E_   48
#define P_   2256
#define NI_  600
#define NJ_  400
#define R_   97
#define BS_  64
#define NNODES 96
#define BP_  (B_*P_)        // 4512
#define KC_  (D_*BS_)       // 49152
#define NE_TOT 1000         // NI_+NJ_

// ---------------------------------------------------------------------------
// index extraction from one-hot maps: mpos[b,m], eassign[b,m]
__global__ void k_indices(const float* __restrict__ mm, const float* __restrict__ em,
                          int* __restrict__ mpos, int* __restrict__ eas) {
  int bm = blockIdx.x; int b = bm / M_; int m = bm % M_;
  __shared__ int sp, se;
  if (threadIdx.x == 0) { sp = 0; se = 0; }
  __syncthreads();
  for (int l = threadIdx.x; l < L_; l += blockDim.x)
    if (mm[(size_t)(b*M_+m)*L_ + l] > 0.5f) sp = l;
  for (int e = threadIdx.x; e < E_; e += blockDim.x)
    if (em[(size_t)(b*E_+e)*M_ + m] > 0.5f) se = e;
  __syncthreads();
  if (threadIdx.x == 0) { mpos[bm] = sp; eas[bm] = se; }
}

// mention[b,m,:] = exp(context[b, mpos, :])
__global__ void k_mention(const float* __restrict__ ctx, const int* __restrict__ mpos,
                          float* __restrict__ mention) {
  int bm = blockIdx.x; int b = bm / M_;
  int l = mpos[bm];
  for (int d = threadIdx.x; d < D_; d += blockDim.x)
    mention[(size_t)bm*D_ + d] = expf(ctx[((size_t)b*L_ + l)*D_ + d]);
}

// entity[b,e,:] = log(sum_{m: eas==e} mention[b,m,:]), 0 if empty
__global__ void k_entity(const float* __restrict__ mention, const int* __restrict__ eas,
                         float* __restrict__ entity) {
  int be = blockIdx.x; int b = be / E_; int e = be % E_;
  for (int d = threadIdx.x; d < D_; d += blockDim.x) {
    float s = 0.f; int cnt = 0;
    for (int m = 0; m < M_; m++)
      if (eas[b*M_+m] == e) { s += mention[(size_t)(b*M_+m)*D_ + d]; cnt++; }
    entity[(size_t)be*D_ + d] = cnt ? logf(s) : 0.f;
  }
}

// ent_att[b,h,e,:] = (1/cnt_e) * sum over mentions of e: attention[b,h,mpos,:]
__global__ void k_entatt(const float* __restrict__ att, const int* __restrict__ mpos,
                         const int* __restrict__ eas, float* __restrict__ ea) {
  int id = blockIdx.x; int e = id % E_; int bh = id / E_; int b = bh / H_;
  __shared__ int sml[M_]; __shared__ int scnt;
  if (threadIdx.x == 0) scnt = 0;
  __syncthreads();
  if (threadIdx.x < M_ && eas[b*M_ + threadIdx.x] == e) {
    int slot = atomicAdd(&scnt, 1); sml[slot] = mpos[b*M_ + threadIdx.x];
  }
  __syncthreads();
  int cnt = scnt;
  float scale = 1.f / ((float)cnt + 1e-20f);
  const float* abase = att + (size_t)bh * L_ * L_;
  for (int l = threadIdx.x; l < L_; l += blockDim.x) {
    float s = 0.f;
    for (int i = 0; i < cnt; i++) s += abase[(size_t)sml[i]*L_ + l];
    ea[(size_t)id*L_ + l] = s * scale;
  }
}

// ca[b,p,:] = normalize_l( sum_h ea[b,h,h0,:]*ea[b,h,h1,:] ) * ht_mask -> f16
__global__ void k_ca(const float* __restrict__ ea, const int* __restrict__ hts,
                     f16* __restrict__ ca16) {
  int bp = blockIdx.x; int b = bp / P_;
  int h0 = hts[bp*2], h1 = hts[bp*2+1];
  float mask = (h0 + h1) != 0 ? 1.f : 0.f;
  float vals[4]; float part = 0.f;
  for (int i = 0; i < 4; i++) {
    int l = threadIdx.x + i*256;
    float s = 0.f;
    for (int h = 0; h < H_; h++) {
      const float* base = ea + (size_t)((b*H_+h)*E_) * L_;
      s += base[(size_t)h0*L_ + l] * base[(size_t)h1*L_ + l];
    }
    vals[i] = s; part += s;
  }
  __shared__ float red[256];
  red[threadIdx.x] = part; __syncthreads();
  for (int s = 128; s > 0; s >>= 1) {
    if (threadIdx.x < s) red[threadIdx.x] += red[threadIdx.x + s];
    __syncthreads();
  }
  float inv = mask / (red[0] + 1e-20f);
  for (int i = 0; i < 4; i++) {
    int l = threadIdx.x + i*256;
    ca16[(size_t)bp*L_ + l] = (f16)(vals[i] * inv);
  }
}

// transpose+convert: dst[n*Ktot + koff + k] = (n<srcN ? src[k*srcN + n] : 0)
__global__ void k_cvt_t(const float* __restrict__ src, f16* __restrict__ dst,
                        int srcK, int srcN, int dstN, int Ktot, int koff) {
  size_t total = (size_t)dstN * srcK;
  for (size_t idx = (size_t)blockIdx.x*blockDim.x + threadIdx.x; idx < total;
       idx += (size_t)gridDim.x*blockDim.x) {
    int n = (int)(idx / srcK); int k = (int)(idx % srcK);
    float v = (n < srcN) ? src[(size_t)k*srcN + n] : 0.f;
    dst[(size_t)n*Ktot + koff + k] = (f16)v;
  }
}

// A_h/A_t cols [0,768): masked entity gather, f16
__global__ void k_build_A(const float* __restrict__ entity, const int* __restrict__ hts,
                          f16* __restrict__ Ah, f16* __restrict__ At) {
  size_t total = (size_t)BP_ * D_;
  for (size_t idx = (size_t)blockIdx.x*blockDim.x + threadIdx.x; idx < total;
       idx += (size_t)gridDim.x*blockDim.x) {
    int bp = (int)(idx / D_); int d = (int)(idx % D_);
    int b = bp / P_;
    int h0 = hts[bp*2], h1 = hts[bp*2+1];
    float mask = (h0 + h1) != 0 ? 1.f : 0.f;
    Ah[(size_t)bp*1536 + d] = (f16)(entity[((size_t)b*E_ + h0)*D_ + d] * mask);
    At[(size_t)bp*1536 + d] = (f16)(entity[((size_t)b*E_ + h1)*D_ + d] * mask);
  }
}

// ---------------------------------------------------------------------------
// WMMA fragment load: 8 halves at p, 8 halves at p+16 (ISA 16x32 f16 layout)
__device__ __forceinline__ v16h load_frag(const f16* p) {
  v8h lo = *(const v8h*)p;
  v8h hi = *(const v8h*)(p + 16);
  v16h r;
#pragma unroll
  for (int i = 0; i < 8; i++) { r[i] = lo[i]; r[i+8] = hi[i]; }
  return r;
}

#define EP_CI   0
#define EP_TANH 1

// Generic f16 GEMM: one wave computes a 16x64 output tile (1 A frag reused
// against 4 B frags -> 4 wmma per 10 b128 loads). B pre-transposed [N][K].
__global__ void k_gemm16(const f16* __restrict__ A, const f16* __restrict__ Bt,
                         int K, int lda, int rowsPerBatch, size_t bBatchStride,
                         int mode, const float* __restrict__ bias1,
                         const float* __restrict__ bias2,
                         f16* __restrict__ out1, f16* __restrict__ out2,
                         int ldo, int out_off) {
  int mt = blockIdx.x, nt = blockIdx.y;       // nt covers 64 output columns
  int lane = threadIdx.x;
  int r15  = lane & 15, kph = lane >> 4;
  int grow = mt*16 + r15;                     // A row (M)
  const f16* bbase = Bt;
  if (rowsPerBatch) { int b = (mt*16) / rowsPerBatch; bbase += (size_t)b * bBatchStride; }
  const f16* ap  = A     + (size_t)grow*lda + kph*8;
  const f16* bp0 = bbase + (size_t)(nt*64 + r15)*K + kph*8;
  v8f c[4];
#pragma unroll
  for (int s = 0; s < 4; s++) c[s] = (v8f){0.f,0.f,0.f,0.f,0.f,0.f,0.f,0.f};
  for (int k = 0; k < K; k += 32) {
    v16h a = load_frag(ap + k);
#pragma unroll
    for (int s = 0; s < 4; s++) {
      v16h b = load_frag(bp0 + (size_t)s*16*K + k);
      c[s] = __builtin_amdgcn_wmma_f32_16x16x32_f16(false, a, false, b, (short)0, c[s],
                                                    false, false);
    }
  }
  int orow0 = mt*16 + 8*kph;
#pragma unroll
  for (int s = 0; s < 4; s++) {
    int ocol = nt*64 + s*16 + r15;
    if (mode == EP_CI) {
#pragma unroll
      for (int j = 0; j < 8; j++) {
        f16 v = (f16)c[s][j];
        size_t o = (size_t)(orow0 + j)*ldo + out_off + ocol;
        out1[o] = v; out2[o] = v;
      }
    } else {
      float bb = bias1[ocol] + bias2[ocol];
#pragma unroll
      for (int j = 0; j < 8; j++)
        out1[(size_t)(orow0 + j)*ldo + ocol] = (f16)tanhf(c[s][j] + bb);
    }
  }
}

// Classifier GEMM: pred = (outer(h,t) per group) @ W_clas + b_clas.
// Block: 16 rows x 112 cols; A-tiles generated on the fly in LDS; 7 compute waves.
__global__ void k_clas_gemm(const f16* __restrict__ h16, const f16* __restrict__ t16,
                            const f16* __restrict__ WcT, const float* __restrict__ bclas,
                            float* __restrict__ pred) {
  __shared__ __align__(16) f16 Hs[16*64];
  __shared__ __align__(16) f16 Ts[16*64];
  __shared__ __align__(16) f16 As[16*64];
  int mt = blockIdx.x;
  int tid = threadIdx.x;
  int wave = tid >> 5, lane = tid & 31;
  int r15 = lane & 15, kph = lane >> 4;
  v8f c = {0.f,0.f,0.f,0.f,0.f,0.f,0.f,0.f};
  for (int n = 0; n < 12; n++) {
    __syncthreads();
    for (int idx = tid; idx < 1024; idx += 256) {
      int r = idx >> 6, jj = idx & 63;
      Hs[idx] = h16[(size_t)(mt*16 + r)*D_ + n*64 + jj];
      Ts[idx] = t16[(size_t)(mt*16 + r)*D_ + n*64 + jj];
    }
    __syncthreads();
    for (int i = 0; i < 64; i++) {
      for (int idx = tid; idx < 1024; idx += 256) {
        int r = idx >> 6;
        As[idx] = (f16)(Hs[(r << 6) + i] * Ts[idx]);
      }
      __syncthreads();
      if (wave < 7) {
        size_t kbase = (size_t)(n*4096 + i*64);
        int gcol = wave*16 + r15;
        const f16* bp = WcT + (size_t)gcol*KC_ + kbase + kph*8;
        // keep the weight stream one chunk ahead of the XDL pipe
        __builtin_prefetch(bp + 1024, 0, 1);
#pragma unroll
        for (int ks = 0; ks < 2; ks++) {
          v16h a = load_frag(&As[(r15 << 6) + ks*32 + kph*8]);
          v16h b = load_frag(bp + ks*32);
          c = __builtin_amdgcn_wmma_f32_16x16x32_f16(false, a, false, b, (short)0, c,
                                                     false, false);
        }
      }
      __syncthreads();
    }
  }
  if (wave < 7) {
    int ocol  = wave*16 + r15;
    int orow0 = mt*16 + 8*kph;
    if (ocol < R_) {
      float bb = bclas[ocol];
#pragma unroll
      for (int j = 0; j < 8; j++)
        pred[(size_t)(orow0 + j)*R_ + ocol] = c[j] + bb;
    }
  }
}

// ---------------------------------------------------------------------------
__global__ void k_zero(float* __restrict__ p, int n) {
  for (int i = blockIdx.x*blockDim.x + threadIdx.x; i < n; i += gridDim.x*blockDim.x)
    p[i] = 0.f;
}

__global__ void k_gather_edges(const float* __restrict__ pred, const int* __restrict__ eidx,
                               float* __restrict__ e, int NE) {
  int total = B_ * NE * R_;
  for (int idx = blockIdx.x*blockDim.x + threadIdx.x; idx < total; idx += gridDim.x*blockDim.x) {
    int r = idx % R_; int be = idx / R_; int b = be / NE; int i = be % NE;
    int p = eidx[b*NE + i];
    e[(size_t)be*R_ + r] = pred[((size_t)b*P_ + p)*R_ + r];
  }
}

__global__ void k_msg(const float* __restrict__ e, const float* __restrict__ Wd,
                      const float* __restrict__ bd, const int* __restrict__ dst,
                      float* __restrict__ rel_sum, float* __restrict__ cnt, int etype) {
  int edge = blockIdx.x;
  int d = dst[edge];
  int r = threadIdx.x;
  if (r < R_) {
    float s = bd[r];
    for (int k = 0; k < R_; k++) s += e[(size_t)edge*R_ + k] * Wd[k*R_ + r];
    float m = s > 0.f ? s : 0.01f * s;   // leaky_relu, slope 0.01
    atomicAdd(&rel_sum[(d*2 + etype)*R_ + r], m);
  }
  if (threadIdx.x == 0) atomicAdd(&cnt[d*2 + etype], 1.f);
}

__global__ void k_rel_div(const float* __restrict__ rel_sum, const float* __restrict__ cnt,
                          float* __restrict__ rel) {
  int total = NNODES*2*R_;
  for (int idx = blockIdx.x*blockDim.x + threadIdx.x; idx < total; idx += gridDim.x*blockDim.x)
    rel[idx] = rel_sum[idx] / fmaxf(cnt[idx / R_], 1.f);
}

__global__ void k_e2(const float* __restrict__ e, const float* __restrict__ rel,
                     const float* __restrict__ We, const float* __restrict__ be,
                     const int* __restrict__ src, const int* __restrict__ dst,
                     float* __restrict__ temp, int NE, int pos_off) {
  int edge = blockIdx.x;
  int b = edge / NE; int i = edge % NE;
  int s = src[edge], d = dst[edge];
  int r = threadIdx.x;
  if (r < R_) {
    float acc = be[r];
    for (int k = 0; k < R_; k++) {
      acc += e[(size_t)edge*R_ + k] * We[k*R_ + r];
      acc += rel[(s*2 + 0)*R_ + k]  * We[(R_ + k)*R_ + r];
      acc += rel[(d*2 + 1)*R_ + k]  * We[(2*R_ + k)*R_ + r];
    }
    temp[((size_t)b*NE_TOT + pos_off + i)*R_ + r] = acc;
  }
}

// stable argsort ranks: rank[j] = #{jj : key<key_j or (key==key_j and jj<j)}
__global__ void k_rank(const int* __restrict__ inter_idx, const int* __restrict__ intra_idx,
                       int* __restrict__ rank) {
  int b = blockIdx.x;
  __shared__ int keys[NE_TOT];
  for (int j = threadIdx.x; j < NE_TOT; j += blockDim.x)
    keys[j] = j < NI_ ? inter_idx[b*NI_ + j] : intra_idx[b*NJ_ + j - NI_];
  __syncthreads();
  for (int j = threadIdx.x; j < NE_TOT; j += blockDim.x) {
    int kj = keys[j]; int r = 0;
    for (int jj = 0; jj < NE_TOT; jj++) {
      int k2 = keys[jj];
      r += (k2 < kj) || (k2 == kj && jj < j);
    }
    rank[b*NE_TOT + j] = r;
  }
}

__global__ void k_scatter(const float* __restrict__ temp, const int* __restrict__ rank,
                          float* __restrict__ out) {
  int total = B_*NE_TOT*R_;
  for (int idx = blockIdx.x*blockDim.x + threadIdx.x; idx < total; idx += gridDim.x*blockDim.x) {
    int r = idx % R_; int bj = idx / R_; int b = bj / NE_TOT; int j = bj % NE_TOT;
    out[((size_t)b*NE_TOT + rank[b*NE_TOT + j])*R_ + r] = temp[(size_t)bj*R_ + r];
  }
}

// ---------------------------------------------------------------------------
extern "C" void kernel_launch(void* const* d_in, const int* in_sizes, int n_in,
                              void* d_out, int out_size, void* d_ws, size_t ws_size,
                              hipStream_t stream) {
  (void)in_sizes; (void)n_in; (void)out_size; (void)ws_size;
  const float* ctx        = (const float*)d_in[0];
  const float* att        = (const float*)d_in[1];
  const float* mention_mp = (const float*)d_in[2];
  const float* entity_mp  = (const float*)d_in[3];
  const int*   hts        = (const int*)  d_in[4];
  const int*   inter_idx  = (const int*)  d_in[5];
  const int*   intra_idx  = (const int*)  d_in[6];
  const int*   inter_src  = (const int*)  d_in[7];
  const int*   inter_dst  = (const int*)  d_in[8];
  const int*   intra_src  = (const int*)  d_in[9];
  const int*   intra_dst  = (const int*)  d_in[10];
  const float* W_h  = (const float*)d_in[11]; const float* b_h  = (const float*)d_in[12];
  const float* W_t  = (const float*)d_in[13]; const float* b_t  = (const float*)d_in[14];
  const float* W_hc = (const float*)d_in[15]; const float* b_hc = (const float*)d_in[16];
  const float* W_tc = (const float*)d_in[17]; const float* b_tc = (const float*)d_in[18];
  const float* W_clas = (const float*)d_in[19]; const float* b_clas = (const float*)d_in[20];
  const float* Wd_inter = (const float*)d_in[21]; const float* bd_inter = (const float*)d_in[22];
  const float* Wd_intra = (const float*)d_in[23]; const float* bd_intra = (const float*)d_in[24];
  const float* We_inter = (const float*)d_in[25]; const float* be_inter = (const float*)d_in[26];
  const float* We_intra = (const float*)d_in[27]; const float* be_intra = (const float*)d_in[28];
  float* out = (float*)d_out;

  char* ws = (char*)d_ws; size_t off = 0;
  auto alloc = [&](size_t bytes) -> char* {
    char* p = ws + off; off += (bytes + 255) & ~(size_t)255; return p;
  };
  int*   mpos    = (int*)  alloc((size_t)B_*M_*4);
  int*   eas     = (int*)  alloc((size_t)B_*M_*4);
  float* mention = (float*)alloc((size_t)B_*M_*D_*4);
  float* entity  = (float*)alloc((size_t)B_*E_*D_*4);
  float* ea      = (float*)alloc((size_t)B_*H_*E_*L_*4);
  f16*   ca16    = (f16*)  alloc((size_t)BP_*L_*2);
  f16*   ctxT16  = (f16*)  alloc((size_t)B_*D_*L_*2);
  f16*   WhT16   = (f16*)  alloc((size_t)D_*1536*2);
  f16*   WtT16   = (f16*)  alloc((size_t)D_*1536*2);
  f16*   WcT16   = (f16*)  alloc((size_t)112*KC_*2);
  f16*   Ah16    = (f16*)  alloc((size_t)BP_*1536*2);
  f16*   At16    = (f16*)  alloc((size_t)BP_*1536*2);
  f16*   h16     = (f16*)  alloc((size_t)BP_*D_*2);
  f16*   t16     = (f16*)  alloc((size_t)BP_*D_*2);
  float* pred    = (float*)alloc((size_t)BP_*R_*4);
  float* e_inter = (float*)alloc((size_t)B_*NI_*R_*4);
  float* e_intra = (float*)alloc((size_t)B_*NJ_*R_*4);
  float* rel_sum = (float*)alloc((size_t)(NNODES*2*R_ + NNODES*2)*4);
  float* cnt     = rel_sum + NNODES*2*R_;
  float* rel     = (float*)alloc((size_t)NNODES*2*R_*4);
  float* temp    = (float*)alloc((size_t)B_*NE_TOT*R_*4);
  int*   rank    = (int*)  alloc((size_t)B_*NE_TOT*4);

  auto gsz = [](size_t n) { return (unsigned)((n + 255) / 256); };

  // stage 0: zero atomic accumulators
  k_zero<<<gsz(NNODES*2*R_ + NNODES*2), 256, 0, stream>>>(rel_sum, NNODES*2*R_ + NNODES*2);
  // stage 1: one-hot -> indices, pooled mentions/entities
  k_indices<<<B_*M_, 256, 0, stream>>>(mention_mp, entity_mp, mpos, eas);
  k_mention<<<B_*M_, 256, 0, stream>>>(ctx, mpos, mention);
  k_entity <<<B_*E_, 256, 0, stream>>>(mention, eas, entity);
  // stage 2: entity attention + pair context weights
  k_entatt <<<B_*H_*E_, 256, 0, stream>>>(att, mpos, eas, ea);
  k_ca     <<<BP_, 256, 0, stream>>>(ea, hts, ca16);
  // stage 3: f16 transposed operands
  for (int b = 0; b < B_; b++)
    k_cvt_t<<<gsz((size_t)D_*L_), 256, 0, stream>>>(ctx + (size_t)b*L_*D_,
        ctxT16 + (size_t)b*D_*L_, L_, D_, D_, L_, 0);
  k_cvt_t<<<gsz((size_t)D_*D_), 256, 0, stream>>>(W_h,  WhT16, D_, D_, D_, 1536, 0);
  k_cvt_t<<<gsz((size_t)D_*D_), 256, 0, stream>>>(W_hc, WhT16, D_, D_, D_, 1536, D_);
  k_cvt_t<<<gsz((size_t)D_*D_), 256, 0, stream>>>(W_t,  WtT16, D_, D_, D_, 1536, 0);
  k_cvt_t<<<gsz((size_t)D_*D_), 256, 0, stream>>>(W_tc, WtT16, D_, D_, D_, 1536, D_);
  k_cvt_t<<<gsz((size_t)112*KC_), 256, 0, stream>>>(W_clas, WcT16, KC_, R_, 112, KC_, 0);
  // stage 4: build A matrices (entity halves), then ci via WMMA into the ci halves
  k_build_A<<<gsz((size_t)BP_*D_), 256, 0, stream>>>(entity, hts, Ah16, At16);
  {
    dim3 g(BP_/16, D_/64);
    k_gemm16<<<g, 32, 0, stream>>>(ca16, ctxT16, L_, L_, P_, (size_t)D_*L_,
                                   EP_CI, nullptr, nullptr, Ah16, At16, 1536, D_);
  }
  // stage 5: head/tail projections (fused [h|ci] @ [W;Wc] + bias, tanh) via WMMA
  {
    dim3 g(BP_/16, D_/64);
    k_gemm16<<<g, 32, 0, stream>>>(Ah16, WhT16, 1536, 1536, 0, 0,
                                   EP_TANH, b_h, b_hc, h16, nullptr, D_, 0);
    k_gemm16<<<g, 32, 0, stream>>>(At16, WtT16, 1536, 1536, 0, 0,
                                   EP_TANH, b_t, b_tc, t16, nullptr, D_, 0);
  }
  // stage 6: group-bilinear classifier, M=4512 K=49152 N=97 via WMMA
  k_clas_gemm<<<BP_/16, 256, 0, stream>>>(h16, t16, WcT16, b_clas, pred);
  // stage 7: hetero-graph
  k_gather_edges<<<gsz((size_t)B_*NI_*R_), 256, 0, stream>>>(pred, inter_idx, e_inter, NI_);
  k_gather_edges<<<gsz((size_t)B_*NJ_*R_), 256, 0, stream>>>(pred, intra_idx, e_intra, NJ_);
  k_msg<<<B_*NI_, 128, 0, stream>>>(e_inter, Wd_inter, bd_inter, inter_dst, rel_sum, cnt, 0);
  k_msg<<<B_*NJ_, 128, 0, stream>>>(e_intra, Wd_intra, bd_intra, intra_dst, rel_sum, cnt, 1);
  k_rel_div<<<gsz(NNODES*2*R_), 256, 0, stream>>>(rel_sum, cnt, rel);
  k_e2<<<B_*NI_, 128, 0, stream>>>(e_inter, rel, We_inter, be_inter, inter_src, inter_dst,
                                   temp, NI_, 0);
  k_e2<<<B_*NJ_, 128, 0, stream>>>(e_intra, rel, We_intra, be_intra, intra_src, intra_dst,
                                   temp, NJ_, NI_);
  // stage 8: stable reorder by original pair index
  k_rank<<<B_, 256, 0, stream>>>(inter_idx, intra_idx, rank);
  k_scatter<<<gsz((size_t)B_*NE_TOT*R_), 256, 0, stream>>>(temp, rank, out);
}
